// ContinuousA2C_34797825032749
// MI455X (gfx1250) — compile-verified, matrix operand order checked by reading, not run on hardware
//
#include <hip/hip_runtime.h>
#include <hip/hip_fp16.h>

// ---------------------------------------------------------------------------
// Continuous A2C (2x stacked LSTM towers + heads) for MI455X / gfx1250.
// One workgroup per 16-batch-row tile; weights resident in LDS; per-timestep
// fused (x@Wk + h@Wr + b) via v_wmma_f32_16x16x32_f16; recurrence carried in
// LDS. x_{t+1} is register-prefetched during step t so HBM latency overlaps
// the WMMA+gate phases. 8 wave32 per WG (2 per SIMD32); N padded 400->512 so
// every wave owns exactly 4 N-tiles -> zero divergent control flow around
// WMMA (EXEC all-ones, no CFG-induced register copies on the serial chain).
// Towers run concurrently as gridDim.y = 2.
// ---------------------------------------------------------------------------

#define TPB      256   // 8 wave32, 2 per SIMD32
#define NWAVE    8
#define NTILES   25    // real N-tiles: 4H = 400 = 25 x 16
#define NTILES_P 32    // padded N-tiles: uniform 4 per wave
#define NTW      4     // tiles per wave (exact)
#define NP       512   // padded 4H
#define HH       100
#define FOURH    400
#define TT       1024
#define BB       512
#define DD       64

typedef __attribute__((ext_vector_type(16))) _Float16 v16h;
typedef __attribute__((ext_vector_type(8)))  _Float16 v8h;
typedef __attribute__((ext_vector_type(8)))  float    v8f;

// Load one 16x32 f16 WMMA fragment (A or B) from an LDS array laid out
// [rows][stride_h halves], rows = M (A) or N (B), inner dim = K.
// CDNA5 16-bit A/B layout: lanes 0-15 -> K 0-7 (v0-3), 16-23 (v4-7);
// lanes 16-31 -> K 8-15, 24-31. Two contiguous ds_load_b128 per lane.
__device__ __forceinline__ v16h load_frag(const _Float16* base, int stride_h,
                                          int ks, int lane) {
  const int r  = lane & 15;
  const int ko = ks * 32 + ((lane & 16) ? 8 : 0);
  const _Float16* p = base + r * stride_h + ko;
  v8h lo = *(const v8h*)(p);
  v8h hi = *(const v8h*)(p + 16);
  v16h f;
#pragma unroll
  for (int i = 0; i < 8; ++i) { f[i] = lo[i]; f[i + 8] = hi[i]; }
  return f;
}

__device__ __forceinline__ float sigmoidf_(float x) {
  return 1.0f / (1.0f + __expf(-x));
}

// One LSTM layer, both towers at once (blockIdx.y selects tower).
// InT: input element type (float for raw x, _Float16 for hidden sequences).
// KIN: true input width (64 or 100). KINP: K padded to multiple of 32.
// SEQ_OUT: store full [B,T,H] f16 sequence vs only final [B,H] f32 state.
template <typename InT, int KIN, int KINP, bool SEQ_OUT>
__global__ __launch_bounds__(TPB) void lstm_layer2x(
    const InT* __restrict__ xin0, const InT* __restrict__ xin1,
    const float* __restrict__ wk0, const float* __restrict__ wr0,
    const float* __restrict__ bb0,
    const float* __restrict__ wk1, const float* __restrict__ wr1,
    const float* __restrict__ bb1,
    _Float16* __restrict__ seq0, _Float16* __restrict__ seq1,
    float* __restrict__ last0, float* __restrict__ last1) {
  constexpr int SIN    = KINP + 8;   // A_x / Wk LDS row stride (halves)
  constexpr int SREC   = 136;        // A_h / Wr LDS row stride (K=100 -> 128)
  constexpr int KS_IN  = KINP / 32;  // WMMA k-steps over input part
  constexpr int KS_REC = 4;          // WMMA k-steps over recurrent part
  constexpr int ZS     = 416;        // z staging row stride (floats)
  constexpr int CS     = 104;        // c state row stride (floats)
  constexpr int NPF    = (16 * KIN + TPB - 1) / TPB;  // prefetch regs/thread

  const int tower = blockIdx.y;
  const InT*   xin   = tower ? xin1 : xin0;
  const float* wk    = tower ? wk1 : wk0;
  const float* wr    = tower ? wr1 : wr0;
  const float* biasp = tower ? bb1 : bb0;
  _Float16* seq  = tower ? seq1 : seq0;
  float*    last = tower ? last1 : last0;

  extern __shared__ __align__(16) char smem_raw[];
  _Float16* wkL = (_Float16*)smem_raw;          // [512][SIN]  input weights^T
  _Float16* wrL = wkL + NP * SIN;               // [512][136]  recur weights^T
  _Float16* axL = wrL + NP * SREC;              // [16][SIN]   x_t tile (f16)
  _Float16* ahL = axL + 16 * SIN;               // [16][136]   h_t tile (f16)
  float*    zL  = (float*)(ahL + 16 * SREC);    // [16][416]   pre-activations
  float*    cL  = zL + 16 * ZS;                 // [16][104]   cell state

  const int tid  = threadIdx.x;
  const int lane = tid & 31;
  const int wave = tid >> 5;
  const int b0   = blockIdx.x * 16;

  // ---- init: zero LDS (covers all zero-padded K rows / N cols) ----
  for (int e = tid; e < NP * SIN;  e += TPB) wkL[e] = (_Float16)0.f;
  for (int e = tid; e < NP * SREC; e += TPB) wrL[e] = (_Float16)0.f;
  for (int e = tid; e < 16 * SIN;  e += TPB) axL[e] = (_Float16)0.f;
  for (int e = tid; e < 16 * SREC; e += TPB) ahL[e] = (_Float16)0.f;
  for (int e = tid; e < 16 * CS;   e += TPB) cL[e]  = 0.f;
  __syncthreads();

  // ---- transpose-convert weights into LDS as [N][K] f16 (read once),
  //      and stage x_0 tile; N cols 400..511 stay zero ----
  for (int e = tid; e < KIN * FOURH; e += TPB) {
    int k = e / FOURH, n = e - k * FOURH;
    wkL[n * SIN + k] = (_Float16)wk[e];
  }
  for (int e = tid; e < HH * FOURH; e += TPB) {
    int k = e / FOURH, n = e - k * FOURH;
    wrL[n * SREC + k] = (_Float16)wr[e];
  }
  for (int e = tid; e < 16 * KIN; e += TPB) {
    int r = e / KIN, c = e - r * KIN;
    axL[r * SIN + c] = (_Float16)(float)xin[((size_t)(b0 + r) * TT) * KIN + c];
  }
  __syncthreads();

  // ---- per-lane bias registers (index-clamped for padded tiles) ----
  float bv[NTW];
#pragma unroll
  for (int i = 0; i < NTW; ++i) {
    int bi = (wave * NTW + i) * 16 + (lane & 15);
    bi = bi > FOURH - 1 ? FOURH - 1 : bi;  // clamp, value unused for pads
    bv[i] = biasp[bi];
  }

  // ---- timestep recurrence ----
  for (int t = 0; t < TT; ++t) {
    // register-prefetch x_{t+1}: independent of this step's WMMA phase, so
    // the global-load latency overlaps it; committed to LDS in gate phase.
    float px[NPF];
    if (t < TT - 1) {
#pragma unroll
      for (int i = 0; i < NPF; ++i) {
        const int e = tid + i * TPB;
        if (e < 16 * KIN) {
          const int r = e / KIN, c = e - r * KIN;
          px[i] = (float)xin[((size_t)(b0 + r) * TT + (t + 1)) * KIN + c];
        }
      }
    }

    // z = x_t @ Wk + h @ Wr + b  via WMMA; wave owns tiles wave*4 .. wave*4+3
    // (uniform, unguarded: EXEC stays all-ones through the whole phase)
    v8f acc[NTW];
#pragma unroll
    for (int i = 0; i < NTW; ++i)
#pragma unroll
      for (int r = 0; r < 8; ++r) acc[i][r] = bv[i];
#pragma unroll
    for (int ks = 0; ks < KS_IN; ++ks) {
      v16h a = load_frag(axL, SIN, ks, lane);
#pragma unroll
      for (int i = 0; i < NTW; ++i) {
        const int nt = wave * NTW + i;
        v16h bf = load_frag(wkL + nt * 16 * SIN, SIN, ks, lane);
        acc[i] = __builtin_amdgcn_wmma_f32_16x16x32_f16(
            false, a, false, bf, (short)0, acc[i], false, false);
      }
    }
#pragma unroll
    for (int ks = 0; ks < KS_REC; ++ks) {
      v16h a = load_frag(ahL, SREC, ks, lane);
#pragma unroll
      for (int i = 0; i < NTW; ++i) {
        const int nt = wave * NTW + i;
        v16h bf = load_frag(wrL + nt * 16 * SREC, SREC, ks, lane);
        acc[i] = __builtin_amdgcn_wmma_f32_16x16x32_f16(
            false, a, false, bf, (short)0, acc[i], false, false);
      }
    }
    // stage z to LDS (C layout: vgpr r -> row r (lanes 0-15) / r+8 (16-31)).
    // Padded tiles (col >= 400) are redirected branch-free into the z row
    // padding (cols 400..415), which the gate phase never reads.
    {
      const int rbase = (lane & 16) ? 8 : 0;
#pragma unroll
      for (int i = 0; i < NTW; ++i) {
        int col = (wave * NTW + i) * 16 + (lane & 15);
        col = col > ZS - 1 ? ZS - 1 : col;
#pragma unroll
        for (int r = 0; r < 8; ++r) zL[(rbase + r) * ZS + col] = acc[i][r];
      }
    }
    __syncthreads();  // all WMMA reads of axL/ahL and z writes complete

    // gates: i,f,o = sigmoid; g = relu; c = f*c + i*g; h = o*relu(c)
    for (int e = tid; e < 16 * HH; e += TPB) {
      const int r = e / HH, j = e - r * HH;
      float zi = zL[r * ZS + j];
      float zf = zL[r * ZS + j + HH];
      float zg = zL[r * ZS + j + 2 * HH];
      float zo = zL[r * ZS + j + 3 * HH];
      float ig = sigmoidf_(zi);
      float fg = sigmoidf_(zf);
      float gg = fmaxf(zg, 0.f);
      float og = sigmoidf_(zo);
      float c  = fg * cL[r * CS + j] + ig * gg;
      cL[r * CS + j] = c;
      float h = og * fmaxf(c, 0.f);
      ahL[r * SREC + j] = (_Float16)h;
      if (SEQ_OUT) {
        seq[((size_t)(b0 + r) * TT + t) * HH + j] = (_Float16)h;
      } else if (t == TT - 1) {
        last[(size_t)(b0 + r) * HH + j] = h;
      }
    }
    // commit prefetched x_{t+1} into the A_x buffer (safe: WMMA phase done)
    if (t < TT - 1) {
#pragma unroll
      for (int i = 0; i < NPF; ++i) {
        const int e = tid + i * TPB;
        if (e < 16 * KIN) {
          const int r = e / KIN, c = e - r * KIN;
          axL[r * SIN + c] = (_Float16)px[i];
        }
      }
    }
    __syncthreads();  // h/c/x staged for step t+1
  }
}

// Dense heads: mu = ah@mu_w+mu_b ; sigma = sigmoid(ah@sg_w+sg_b)+1e-5 ;
// value = ch@v_w+v_b. Tiny (512 x {4,4,1}) — plain VALU kernel.
__global__ __launch_bounds__(256) void heads_kernel(
    const float* __restrict__ ah, const float* __restrict__ ch,
    const float* __restrict__ mu_w, const float* __restrict__ mu_b,
    const float* __restrict__ sg_w, const float* __restrict__ sg_b,
    const float* __restrict__ v_w, const float* __restrict__ v_b,
    float* __restrict__ out) {
  int b = blockIdx.x * blockDim.x + threadIdx.x;
  if (b >= BB) return;
  float mu[4] = {mu_b[0], mu_b[1], mu_b[2], mu_b[3]};
  float sg[4] = {sg_b[0], sg_b[1], sg_b[2], sg_b[3]};
  float v = v_b[0];
  for (int k = 0; k < HH; ++k) {
    float a = ah[b * HH + k];
    float c = ch[b * HH + k];
#pragma unroll
    for (int j = 0; j < 4; ++j) {
      mu[j] += a * mu_w[k * 4 + j];
      sg[j] += a * sg_w[k * 4 + j];
    }
    v += c * v_w[k];
  }
#pragma unroll
  for (int j = 0; j < 4; ++j) {
    out[b * 4 + j]        = mu[j];
    out[2048 + b * 4 + j] = 1.0f / (1.0f + __expf(-sg[j])) + 1e-5f;
  }
  out[4096 + b] = v;
}

extern "C" void kernel_launch(void* const* d_in, const int* in_sizes, int n_in,
                              void* d_out, int out_size, void* d_ws,
                              size_t ws_size, hipStream_t stream) {
  (void)in_sizes; (void)n_in; (void)out_size; (void)ws_size;
  const float* x    = (const float*)d_in[0];
  const float* a1k  = (const float*)d_in[1];
  const float* a1r  = (const float*)d_in[2];
  const float* a1b  = (const float*)d_in[3];
  const float* a2k  = (const float*)d_in[4];
  const float* a2r  = (const float*)d_in[5];
  const float* a2b  = (const float*)d_in[6];
  const float* c1k  = (const float*)d_in[7];
  const float* c1r  = (const float*)d_in[8];
  const float* c1b  = (const float*)d_in[9];
  const float* c2k  = (const float*)d_in[10];
  const float* c2r  = (const float*)d_in[11];
  const float* c2b  = (const float*)d_in[12];
  const float* mu_w = (const float*)d_in[13];
  const float* mu_b = (const float*)d_in[14];
  const float* sg_w = (const float*)d_in[15];
  const float* sg_b = (const float*)d_in[16];
  const float* v_w  = (const float*)d_in[17];
  const float* v_b  = (const float*)d_in[18];
  float* out = (float*)d_out;

  // workspace: two f16 hidden sequences [B,T,H] + two f32 final states [B,H]
  char* ws = (char*)d_ws;
  const size_t hseq_bytes = (size_t)BB * TT * HH * sizeof(_Float16);
  _Float16* hseqA = (_Float16*)ws;
  _Float16* hseqC = (_Float16*)(ws + hseq_bytes);
  float* ah = (float*)(ws + 2 * hseq_bytes);
  float* ch = ah + (size_t)BB * HH;

  // dynamic LDS footprints (must mirror lstm_layer2x layout)
  auto smem_of = [](int kinp) -> size_t {
    const size_t sin = (size_t)kinp + 8;
    return (size_t)NP * sin * 2 + (size_t)NP * 136 * 2 +
           16 * sin * 2 + 16 * 136 * 2 +
           (size_t)16 * 416 * 4 + (size_t)16 * 104 * 4;
  };
  const size_t smem1 = smem_of(64);    // ~247 KB
  const size_t smem2 = smem_of(128);   // 320,512 B (<= 320 KB WGP LDS)

  hipFuncSetAttribute(
      reinterpret_cast<const void*>(lstm_layer2x<float, DD, 64, true>),
      hipFuncAttributeMaxDynamicSharedMemorySize, (int)smem1);
  hipFuncSetAttribute(
      reinterpret_cast<const void*>(lstm_layer2x<_Float16, HH, 128, false>),
      hipFuncAttributeMaxDynamicSharedMemorySize, (int)smem2);

  dim3 grid(BB / 16, 2);  // 32 batch tiles x 2 towers

  // layer 1 (D=64 -> H), both towers, full sequences out (f16)
  lstm_layer2x<float, DD, 64, true><<<grid, TPB, smem1, stream>>>(
      x, x, a1k, a1r, a1b, c1k, c1r, c1b, hseqA, hseqC, nullptr, nullptr);
  // layer 2 (H -> H), both towers, final state only (f32)
  lstm_layer2x<_Float16, HH, 128, false><<<grid, TPB, smem2, stream>>>(
      hseqA, hseqC, a2k, a2r, a2b, c2k, c2r, c2b, nullptr, nullptr, ah, ch);
  // heads
  heads_kernel<<<dim3(2), dim3(256), 0, stream>>>(ah, ch, mu_w, mu_b, sg_w,
                                                  sg_b, v_w, v_b, out);
}
